// SelfAttention_21406117003506
// MI455X (gfx1250) — compile-verified, matrix operand order checked by reading, not run on hardware
//
#include <hip/hip_runtime.h>
#include <hip/hip_bf16.h>
#include <cstdint>
#include <cstddef>

// ---------------------------------------------------------------------------
// Types / helpers
// ---------------------------------------------------------------------------
typedef unsigned short u16;
typedef __attribute__((ext_vector_type(16))) __bf16 v16bf;
typedef __attribute__((ext_vector_type(8)))  float  v8f;
typedef __attribute__((ext_vector_type(4)))  int    v4i;

#define AS1 __attribute__((address_space(1)))
#define AS3 __attribute__((address_space(3)))

union F8 { v8f v; float f[8]; };
struct BFrag { union { v16bf v; uint4 q[2]; u16 u[16]; }; };
union VReg { uint4 q; u16 u[8]; };

static __device__ __forceinline__ u16 f32_to_bf16(float x) {
  union { float f; unsigned u; } c; c.f = x;
  unsigned r = c.u + 0x7FFFu + ((c.u >> 16) & 1u);   // round-to-nearest-even
  return (u16)(r >> 16);
}
static __device__ __forceinline__ v8f zero8() {
  v8f z = {0.f,0.f,0.f,0.f,0.f,0.f,0.f,0.f}; return z;
}
static __device__ __forceinline__ v8f wmma_bf16(v16bf a, v16bf b, v8f c) {
  // D = A(16x32 bf16) * B(32x16 bf16) + C(16x16 f32)
  return __builtin_amdgcn_wmma_f32_16x16x32_bf16(false, a, false, b, (short)0, c,
                                                 false, false);
}
static __device__ __forceinline__ void wave_fence() {
#if __has_builtin(__builtin_amdgcn_wave_barrier)
  __builtin_amdgcn_wave_barrier();
#endif
  __asm__ volatile("" ::: "memory");
}

// ---- async global->LDS copy (CDNA5 GLOBAL_LOAD_ASYNC_TO_LDS_B128) ----------
#if __has_builtin(__builtin_amdgcn_global_load_async_to_lds_b128)
#define HAVE_ASYNC_LDS 1
#else
#define HAVE_ASYNC_LDS 0
#endif

static __device__ __forceinline__ void async_copy16(void* lds, const void* g) {
#if HAVE_ASYNC_LDS
  // (v4i AS1*, v4i AS3*, imm offset, imm cpol). Integer round-trip casts:
  // flat/global share bit patterns; low 32 bits of a generic LDS address are
  // the LDS byte offset (ISA 10.2 aperture rules).
  AS1 v4i* gp = (AS1 v4i*)(uintptr_t)g;
  AS3 v4i* lp = (AS3 v4i*)(unsigned)(uintptr_t)lds;
  __builtin_amdgcn_global_load_async_to_lds_b128(gp, lp, 0, 0);
#else
  *(uint4*)lds = *(const uint4*)g;
#endif
}
static __device__ __forceinline__ void async_wait0() {
#if HAVE_ASYNC_LDS
#if __has_builtin(__builtin_amdgcn_s_wait_asynccnt)
  __builtin_amdgcn_s_wait_asynccnt(0);
#else
  __asm__ volatile("s_wait_asynccnt 0x0" ::: "memory");
#endif
#endif
}

// ---- xor-butterfly within 16-lane rows: v_permlane16_b32 (VALU, no DS) -----
template <int MASK>
static __device__ __forceinline__ float lanexor16f(float v) {
#if __has_builtin(__builtin_amdgcn_permlane16)
  int s0, s1;
  if      (MASK == 1) { s0 = (int)0x67452301; s1 = (int)0xEFCDAB89; }
  else if (MASK == 2) { s0 = (int)0x54761032; s1 = (int)0xDCFE98BA; }
  else if (MASK == 4) { s0 = (int)0x32107654; s1 = (int)0xBA98FEDC; }
  else                { s0 = (int)0xFEDCBA98; s1 = (int)0x76543210; }
  int x = __builtin_bit_cast(int, v);
  int r = __builtin_amdgcn_permlane16(x, x, s0, s1, false, false);
  return __builtin_bit_cast(float, r);
#else
  return __shfl_xor(v, MASK, 32);
#endif
}

#define BATCH   2
#define NSEQ    2048
#define DIMM    1024
#define NHEADS  16
#define DHEAD   64
#define ROWS    (BATCH * NSEQ)          // 4096
#define ATTN_SCALE 0.125f               // 64^-0.5

// ---------------------------------------------------------------------------
// fp32 -> bf16 convert (weights)
// ---------------------------------------------------------------------------
__global__ void __launch_bounds__(256) cvt_bf16_kernel(const float* __restrict__ in,
                                                       u16* __restrict__ out, int n) {
  int i = blockIdx.x * 256 + threadIdx.x;
  if (i < n) out[i] = f32_to_bf16(in[i]);
}

// ---------------------------------------------------------------------------
// LayerNorm over dim=1024, one row per block (256 threads, 4 elems/thread)
// ---------------------------------------------------------------------------
template <int OUT_BF16>
__global__ void __launch_bounds__(256) ln_kernel(const float* __restrict__ x,
                                                 const float* __restrict__ g,
                                                 const float* __restrict__ bb,
                                                 void* __restrict__ outp) {
  const int row = blockIdx.x;
  const int tid = threadIdx.x;
  const float* xr = x + (size_t)row * DIMM;
  float v0[4];
  float s = 0.f, s2 = 0.f;
#pragma unroll
  for (int j = 0; j < 4; ++j) {
    float v = xr[tid + 256 * j];
    v0[j] = v; s += v; s2 += v * v;
  }
#pragma unroll
  for (int m = 16; m >= 1; m >>= 1) {
    s  += __shfl_xor(s,  m, 32);
    s2 += __shfl_xor(s2, m, 32);
  }
  __shared__ float sred[16];
  const int w = tid >> 5, ln = tid & 31;
  if (ln == 0) { sred[w] = s; sred[8 + w] = s2; }
  __syncthreads();
  float ts = 0.f, ts2 = 0.f;
#pragma unroll
  for (int i = 0; i < 8; ++i) { ts += sred[i]; ts2 += sred[8 + i]; }
  const float mean = ts * (1.f / DIMM);
  const float var  = ts2 * (1.f / DIMM) - mean * mean;
  const float rstd = rsqrtf(var + 1e-5f);
#pragma unroll
  for (int j = 0; j < 4; ++j) {
    int c = tid + 256 * j;
    float y = (v0[j] - mean) * rstd * g[c] + bb[c];
    if (OUT_BF16) ((u16*)outp)[(size_t)row * DIMM + c] = f32_to_bf16(y);
    else          ((float*)outp)[(size_t)row * DIMM + c] = y;
  }
}

// ---------------------------------------------------------------------------
// bf16 GEMM: C[M,N] = A[M,K] * B[K,N], fp32 accumulate via v_wmma.
// Block tile 128x64, BK=32, 8 waves, each wave computes 32x32 (4 WMMA tiles).
// Double-buffered LDS: async A-copy for tile t+1 overlaps the WMMAs of tile t.
// MODE 0: fp32 out | MODE 1: bf16 q[b,h,n,d] scatter+scale | MODE 2: bf16 out
// ---------------------------------------------------------------------------
template <int MODE>
__global__ void __launch_bounds__(256) gemm_bf16_kernel(const u16* __restrict__ A,
                                                        const u16* __restrict__ Bm,
                                                        void* __restrict__ Out,
                                                        int M, int N, int K, float scale) {
  __shared__ u16 As[2][128][40];   // A tile, row-major  [m][k], pad 32->40
  __shared__ u16 Bs[2][64][40];    // B tile, col-major  [n][k], pad 32->40
  const int tid  = threadIdx.x;
  const int w    = tid >> 5;
  const int lane = tid & 31;
  const int lo16 = lane & 15;
  const int hi   = lane >> 4;
  const int m0 = blockIdx.y * 128;
  const int n0 = blockIdx.x * 64;
  const int mrow = (w & 3) * 32;
  const int ncol = (w >> 2) * 32;

  F8 acc[2][2];
#pragma unroll
  for (int i = 0; i < 2; ++i)
#pragma unroll
    for (int j = 0; j < 2; ++j) acc[i][j].v = zero8();

  const int kb = hi * 8;
  const int arow = tid >> 1;            // staging coords (constant per thread)
  const int aseg = (tid & 1) * 16;
  const int bkk  = tid >> 3;
  const int bcs  = (tid & 7) * 8;
  const int T = K >> 5;                 // number of 32-wide K tiles

  { // prologue: stage tile 0 into buffer 0
    const u16* srcA = A + (size_t)(m0 + arow) * K + aseg;
    async_copy16(&As[0][arow][aseg],     srcA);
    async_copy16(&As[0][arow][aseg + 8], srcA + 8);
    VReg t0;
    t0.q = *(const uint4*)(Bm + (size_t)bkk * N + n0 + bcs);
#pragma unroll
    for (int j = 0; j < 8; ++j) Bs[0][bcs + j][bkk] = t0.u[j];
    async_wait0();
    __syncthreads();
  }

  for (int t = 0; t < T; ++t) {
    const int cur = t & 1;
    const int nxt = cur ^ 1;
    const bool hn = (t + 1 < T);
    VReg bnext;
    if (hn) { // issue next tile's copies before computing this tile
      const int k1 = (t + 1) << 5;
      const u16* srcA = A + (size_t)(m0 + arow) * K + k1 + aseg;
      async_copy16(&As[nxt][arow][aseg],     srcA);
      async_copy16(&As[nxt][arow][aseg + 8], srcA + 8);
      bnext.q = *(const uint4*)(Bm + (size_t)(k1 + bkk) * N + n0 + bcs);
      if (t + 2 < T) {
        const int k2 = (t + 2) << 5;
        __builtin_prefetch(A  + (size_t)(m0 + arow) * K + k2 + aseg, 0, 3);
        __builtin_prefetch(Bm + (size_t)(k2 + bkk) * N + n0 + bcs, 0, 3);
      }
    }

    // compute on buffer `cur` while next tile streams in
    BFrag af[2], bfr[2];
#pragma unroll
    for (int mi = 0; mi < 2; ++mi) {
      const u16* p = &As[cur][mrow + 16 * mi + lo16][0];
      af[mi].q[0] = *(const uint4*)(p + kb);
      af[mi].q[1] = *(const uint4*)(p + kb + 16);
    }
#pragma unroll
    for (int ni = 0; ni < 2; ++ni) {
      const u16* p = &Bs[cur][ncol + 16 * ni + lo16][0];
      bfr[ni].q[0] = *(const uint4*)(p + kb);
      bfr[ni].q[1] = *(const uint4*)(p + kb + 16);
    }
#pragma unroll
    for (int mi = 0; mi < 2; ++mi)
#pragma unroll
      for (int ni = 0; ni < 2; ++ni)
        acc[mi][ni].v = wmma_bf16(af[mi].v, bfr[ni].v, acc[mi][ni].v);

    if (hn) { // finish staging next tile, then hand off buffers
#pragma unroll
      for (int j = 0; j < 8; ++j) Bs[nxt][bcs + j][bkk] = bnext.u[j];
      async_wait0();
      __syncthreads();
    }
  }

  // Epilogue. C layout: element r of the v8f lives at (Mrow = r + 8*hi, Ncol = lo16).
#pragma unroll
  for (int mi = 0; mi < 2; ++mi)
#pragma unroll
    for (int ni = 0; ni < 2; ++ni)
#pragma unroll
      for (int r = 0; r < 8; ++r) {
        int gm = m0 + mrow + 16 * mi + r + 8 * hi;
        int gn = n0 + ncol + 16 * ni + lo16;
        float val = acc[mi][ni].f[r] * scale;
        if (MODE == 0) {
          ((float*)Out)[(size_t)gm * N + gn] = val;
        } else if (MODE == 1) {
          int h = gn >> 6, dd = gn & 63;
          int bt = gm >> 11, i = gm & (NSEQ - 1);
          ((u16*)Out)[(((size_t)bt * NHEADS + h) * NSEQ + i) * DHEAD + dd] =
              f32_to_bf16(val);
        } else {
          ((u16*)Out)[(size_t)gm * N + gn] = f32_to_bf16(val);
        }
      }
}

// ---------------------------------------------------------------------------
// Flash attention (MQA): grid (n/128, heads, batch), block 256 (8 waves).
// Each wave owns 16 queries; all waves share double-buffered 32-key K/V tiles;
// the async K-copy for the next key block overlaps this block's WMMAs+softmax.
// ---------------------------------------------------------------------------
__global__ void __launch_bounds__(256) attn_kernel(const u16* __restrict__ q,
                                                   const u16* __restrict__ kv,
                                                   u16* __restrict__ o) {
  __shared__ u16 Ks[2][32][72];     // [key][d]   pad 64->72
  __shared__ u16 Vs[2][64][40];     // [d][key]   pad 32->40
  __shared__ u16 Ps[8][16][40];     // per-wave P staging [qrow][key] pad 32->40

  const int tid  = threadIdx.x;
  const int w    = tid >> 5;
  const int lane = tid & 31;
  const int lo16 = lane & 15;
  const int hi   = lane >> 4;
  const int kb   = hi * 8;
  const int head  = blockIdx.y;
  const int batch = blockIdx.z;
  const int q0 = blockIdx.x * 128 + w * 16;

  // Q fragments (A-layout): row = lo16, K-halves per hi. d=0..31 and d=32..63.
  BFrag a0, a1;
  {
    const u16* qbp =
        q + (((size_t)batch * NHEADS + head) * NSEQ + q0 + lo16) * DHEAD;
    a0.q[0] = *(const uint4*)(qbp + kb);
    a0.q[1] = *(const uint4*)(qbp + kb + 16);
    a1.q[0] = *(const uint4*)(qbp + 32 + kb);
    a1.q[1] = *(const uint4*)(qbp + 48 + kb);
  }

  float mrow[8], lrow[8];
  F8 accd[4];
#pragma unroll
  for (int r = 0; r < 8; ++r) { mrow[r] = -1e30f; lrow[r] = 0.f; }
#pragma unroll
  for (int d = 0; d < 4; ++d) accd[d].v = zero8();

  const u16* kvb = kv + (size_t)batch * NSEQ * 128;
  const int ski  = tid >> 3;            // staging key index 0..31
  const int sseg = (tid & 7) * 8;       // staging d offset 0..56
  const int TB = NSEQ / 32;             // 64 key blocks

  { // prologue: stage key block 0 into buffer 0
    const u16* src = kvb + (size_t)ski * 128;
    async_copy16(&Ks[0][ski][sseg], src + sseg);
    VReg tv;
    tv.q = *(const uint4*)(src + 64 + sseg);
#pragma unroll
    for (int j = 0; j < 8; ++j) Vs[0][sseg + j][ski] = tv.u[j];
    async_wait0();
    __syncthreads();
  }

  for (int t = 0; t < TB; ++t) {
    const int cur = t & 1;
    const int nxt = cur ^ 1;
    const bool hn = (t + 1 < TB);
    VReg tv;
    if (hn) { // issue next key block's copies up front
      const u16* src = kvb + (size_t)((t + 1) * 32 + ski) * 128;
      async_copy16(&Ks[nxt][ski][sseg], src + sseg);
      tv.q = *(const uint4*)(src + 64 + sseg);
      if (t + 2 < TB)
        __builtin_prefetch(kvb + (size_t)((t + 2) * 32 + ski) * 128 + sseg, 0, 3);
    }

    // --- sim = Q * K^T for 32 keys (two 16x16 tiles, K-dim 64 split in two) ---
    BFrag b0l, b0h, b1l, b1h;
    {
      const u16* p0 = &Ks[cur][lo16][0];
      const u16* p1 = &Ks[cur][16 + lo16][0];
      b0l.q[0] = *(const uint4*)(p0 + kb);      b0l.q[1] = *(const uint4*)(p0 + kb + 16);
      b0h.q[0] = *(const uint4*)(p0 + 32 + kb); b0h.q[1] = *(const uint4*)(p0 + 48 + kb);
      b1l.q[0] = *(const uint4*)(p1 + kb);      b1l.q[1] = *(const uint4*)(p1 + kb + 16);
      b1h.q[0] = *(const uint4*)(p1 + 32 + kb); b1h.q[1] = *(const uint4*)(p1 + 48 + kb);
    }
    F8 s0, s1; s0.v = zero8(); s1.v = zero8();
    s0.v = wmma_bf16(a0.v, b0l.v, s0.v);
    s0.v = wmma_bf16(a1.v, b0h.v, s0.v);
    s1.v = wmma_bf16(a0.v, b1l.v, s1.v);
    s1.v = wmma_bf16(a1.v, b1h.v, s1.v);

    // --- online softmax across the 32 new columns (permlane16 reductions) ---
    float mx[8];
#pragma unroll
    for (int r = 0; r < 8; ++r) mx[r] = fmaxf(s0.f[r], s1.f[r]);
#pragma unroll
    for (int r = 0; r < 8; ++r) mx[r] = fmaxf(mx[r], lanexor16f<1>(mx[r]));
#pragma unroll
    for (int r = 0; r < 8; ++r) mx[r] = fmaxf(mx[r], lanexor16f<2>(mx[r]));
#pragma unroll
    for (int r = 0; r < 8; ++r) mx[r] = fmaxf(mx[r], lanexor16f<4>(mx[r]));
#pragma unroll
    for (int r = 0; r < 8; ++r) mx[r] = fmaxf(mx[r], lanexor16f<8>(mx[r]));

    float p0v[8], p1v[8], sum[8];
#pragma unroll
    for (int r = 0; r < 8; ++r) {
      float mn   = fmaxf(mrow[r], mx[r]);
      float corr = __expf(mrow[r] - mn);
      p0v[r] = __expf(s0.f[r] - mn);
      p1v[r] = __expf(s1.f[r] - mn);
      sum[r] = p0v[r] + p1v[r];
      mrow[r] = mn;
      lrow[r] *= corr;
#pragma unroll
      for (int d = 0; d < 4; ++d) accd[d].f[r] *= corr;
    }
#pragma unroll
    for (int r = 0; r < 8; ++r) sum[r] += lanexor16f<1>(sum[r]);
#pragma unroll
    for (int r = 0; r < 8; ++r) sum[r] += lanexor16f<2>(sum[r]);
#pragma unroll
    for (int r = 0; r < 8; ++r) sum[r] += lanexor16f<4>(sum[r]);
#pragma unroll
    for (int r = 0; r < 8; ++r) sum[r] += lanexor16f<8>(sum[r]);
#pragma unroll
    for (int r = 0; r < 8; ++r) lrow[r] += sum[r];

    // --- re-shape P (C-layout) into A-fragment layout via per-wave LDS ---
#pragma unroll
    for (int r = 0; r < 8; ++r) {
      int prow = r + 8 * hi;
      Ps[w][prow][lo16]      = f32_to_bf16(p0v[r]);
      Ps[w][prow][lo16 + 16] = f32_to_bf16(p1v[r]);
    }
    wave_fence();   // DS is in-order per wave; fence stops compiler reordering
    BFrag pf;
    {
      const u16* p = &Ps[w][lo16][0];
      pf.q[0] = *(const uint4*)(p + kb);
      pf.q[1] = *(const uint4*)(p + kb + 16);
    }

    // --- O += P * V (A 16x32, four 16x16 output tiles along d) ---
#pragma unroll
    for (int d = 0; d < 4; ++d) {
      BFrag bv;
      const u16* p = &Vs[cur][d * 16 + lo16][0];
      bv.q[0] = *(const uint4*)(p + kb);
      bv.q[1] = *(const uint4*)(p + kb + 16);
      accd[d].v = wmma_bf16(pf.v, bv.v, accd[d].v);
    }

    if (hn) { // finish staging next block's V, then hand off buffers
#pragma unroll
      for (int j = 0; j < 8; ++j) Vs[nxt][sseg + j][ski] = tv.u[j];
      async_wait0();
      __syncthreads();
    }
  }

  // epilogue: normalize and write o[b*n, h*64 + d] as bf16
  float inv[8];
#pragma unroll
  for (int r = 0; r < 8; ++r) inv[r] = 1.f / lrow[r];
#pragma unroll
  for (int d = 0; d < 4; ++d)
#pragma unroll
    for (int r = 0; r < 8; ++r) {
      int orow = batch * NSEQ + q0 + r + 8 * hi;
      int ocol = head * DHEAD + d * 16 + lo16;
      o[(size_t)orow * (NHEADS * DHEAD) + ocol] = f32_to_bf16(accd[d].f[r] * inv[r]);
    }
}

// ---------------------------------------------------------------------------
// launch
// ---------------------------------------------------------------------------
extern "C" void kernel_launch(void* const* d_in, const int* in_sizes, int n_in,
                              void* d_out, int out_size, void* d_ws, size_t ws_size,
                              hipStream_t stream) {
  (void)in_sizes; (void)n_in; (void)out_size; (void)ws_size;
  const float* x     = (const float*)d_in[0];
  const float* ln1_g = (const float*)d_in[1];
  const float* ln1_b = (const float*)d_in[2];
  const float* Wq    = (const float*)d_in[3];
  const float* Wkv   = (const float*)d_in[4];
  const float* Wo    = (const float*)d_in[5];
  const float* ln2_g = (const float*)d_in[6];
  const float* ln2_b = (const float*)d_in[7];
  float* out = (float*)d_out;

  char* ws = (char*)d_ws;
  u16*   xn  = (u16*)(ws);                 // 4096x1024 bf16   8 MB
  u16*   wqb = (u16*)(ws + 8388608);       // 1024x1024 bf16   2 MB
  u16*   wkb = (u16*)(ws + 10485760);      // 1024x128  bf16   256 KB
  u16*   wob = (u16*)(ws + 10747904);      // 1024x1024 bf16   2 MB
  u16*   qb  = (u16*)(ws + 12845056);      // [b,h,n,d] bf16   8 MB
  u16*   kvb = (u16*)(ws + 21233664);      // [b*n,128] bf16   1 MB
  u16*   ob  = (u16*)(ws + 22282240);      // [b*n,1024] bf16  8 MB
  float* yb  = (float*)(ws + 30670848);    // [b*n,1024] f32  16 MB

  // 1) weights -> bf16
  cvt_bf16_kernel<<<(DIMM * DIMM + 255) / 256, 256, 0, stream>>>(Wq, wqb, DIMM * DIMM);
  cvt_bf16_kernel<<<(DIMM * 2 * DHEAD + 255) / 256, 256, 0, stream>>>(Wkv, wkb, DIMM * 2 * DHEAD);
  cvt_bf16_kernel<<<(DIMM * DIMM + 255) / 256, 256, 0, stream>>>(Wo, wob, DIMM * DIMM);

  // 2) LN1 -> bf16 xn
  ln_kernel<1><<<ROWS, 256, 0, stream>>>(x, ln1_g, ln1_b, (void*)xn);

  // 3) q = (xn @ Wq) * SCALE  -> [b,h,n,d] bf16
  gemm_bf16_kernel<1><<<dim3(DIMM / 64, ROWS / 128), 256, 0, stream>>>(
      xn, wqb, (void*)qb, ROWS, DIMM, DIMM, ATTN_SCALE);

  // 4) kv = xn @ Wkv -> [b*n,128] bf16
  gemm_bf16_kernel<2><<<dim3(128 / 64, ROWS / 128), 256, 0, stream>>>(
      xn, wkb, (void*)kvb, ROWS, 128, DIMM, 1.0f);

  // 5) flash attention -> o bf16 [b*n, 1024]
  attn_kernel<<<dim3(NSEQ / 128, NHEADS, BATCH), 256, 0, stream>>>(qb, kvb, ob);

  // 6) y = o @ Wo -> fp32
  gemm_bf16_kernel<0><<<dim3(DIMM / 64, ROWS / 128), 256, 0, stream>>>(
      ob, wob, (void*)yb, ROWS, DIMM, DIMM, 1.0f);

  // 7) LN2 -> out fp32
  ln_kernel<0><<<ROWS, 256, 0, stream>>>(yb, ln2_g, ln2_b, (void*)out);
}
